// GlobalPointer_43989055045822
// MI455X (gfx1250) — matrix operand order repaired and checked by reading, not compiled, not run on hardware
//
#include <hip/hip_runtime.h>
#include <hip/hip_bf16.h>

typedef __attribute__((ext_vector_type(16))) _Float16 v16h;
typedef __attribute__((ext_vector_type(8)))  _Float16 v8h;
typedef __attribute__((ext_vector_type(4)))  _Float16 v4h;
typedef __attribute__((ext_vector_type(8)))  float    v8f;
typedef __attribute__((ext_vector_type(4)))  unsigned int u32x4;
typedef __attribute__((ext_vector_type(4)))  int          i32x4;
typedef __attribute__((ext_vector_type(8)))  int          i32x8;

#define N_HEADS   12
#define HEAD_SZ   64
#define SEQ       1024
#define BATCH     8
#define DIM       1024
#define NCOLS     1536          // HEADS * 2 * HEAD_SIZE
#define NEG_BIG   1000000000000.0f
#define LN10000_OVER_HS (9.2103403719761836f / 64.0f)

union FragU { v16h v; v8h h[2]; };

#define LDS_OFFSET(p) ((unsigned)(unsigned long long)(const void*)(p))

// ---------------------------------------------------------------------------
// TDM: 2D tile load Global -> LDS, rows x 64 halves, contiguous source,
// LDS padded +16B per 128B row (pad_interval=32 dw, pad_amount=4 dw)
// -> effective LDS row stride = 72 halves (144 B, 16B aligned rows).
// D# layout per CDNA5 ISA.
// ---------------------------------------------------------------------------
__device__ __forceinline__ void tdm_load_2d_rows64(const void* gsrc,
                                                   unsigned lds_byte_off,
                                                   unsigned rows)
{
    const unsigned long long ga = (unsigned long long)gsrc;
    u32x4 g0;
    g0[0] = 1u;                                      // count=1 (valid), user mode
    g0[1] = lds_byte_off;                            // lds_addr
    g0[2] = (unsigned)(ga & 0xFFFFFFFFu);            // global_addr[31:0]
    g0[3] = (unsigned)((ga >> 32) & 0x1FFFFFFu)      // global_addr[56:32]
          | (2u << 30);                              // type = 2 ("image")
    i32x8 g1;
    g1[0] = (1 << 16)                                // data_size = 2 bytes
          | (1 << 20)                                // pad_enable
          | (4 << 22)                                // pad_interval: 32 DWORDs (128B)
          | (3 << 25);                               // pad_amount: 4 DWORDs (16B)
    g1[1] = (int)(64u << 16);                        // tensor_dim0 = 64 elems (bits 79:48)
    g1[2] = (int)(rows << 16);                       // tensor_dim1 lo16 (bits 95:80)
    g1[3] = (int)(64u << 16);                        // tensor_dim1 hi=0 | tile_dim0 = 64
    g1[4] = (int)rows;                               // tile_dim1 = rows | tile_dim2 = 0
    g1[5] = 64;                                      // tensor_dim0_stride = 64 elems
    g1[6] = 0;                                       // stride hi | dim1_stride lo
    g1[7] = 0;
    const i32x4 z4 = {0, 0, 0, 0};
#if defined(__clang_major__) && (__clang_major__ >= 23)
    const i32x8 z8 = {0, 0, 0, 0, 0, 0, 0, 0};
    __builtin_amdgcn_tensor_load_to_lds(g0, g1, z4, z4, z8, 0);
#else
    __builtin_amdgcn_tensor_load_to_lds(g0, g1, z4, z4, 0);
#endif
}

// ---------------------------------------------------------------------------
// Kernel 1: proj = inputs @ W + bias ; RoPE ; scatter to q_ws / k_ws (f16)
// A tile 128x32 (f32->f16 in LDS), B tile 32x128 stored transposed [n][k].
// Each block: 128(M) x 128(N); 8 waves -> wave tile 64x32 (4x2 WMMA subtiles).
// ---------------------------------------------------------------------------
__global__ __launch_bounds__(256)
void proj_rope_kernel(const float* __restrict__ inp,   // [8192][1024]
                      const float* __restrict__ W,     // [1024][1536]
                      const float* __restrict__ bias,  // [1536]
                      _Float16* __restrict__ qws,      // [96][1024][64]
                      _Float16* __restrict__ kws)      // [96][1024][64]
{
    constexpr int SA = 40;                       // padded half-stride (80 B rows)
    __shared__ _Float16 As[128 * SA];            // 10240 B
    __shared__ _Float16 Bs[128 * SA];            // 10240 B  ([n][k] transposed)
    __shared__ _Float16 Tr[8 * 256];             // per-wave 16x16 transpose pad

    const int t    = threadIdx.x;
    const int lane = t & 31;
    const int wave = t >> 5;
    const int r    = lane & 15;
    const int hi   = lane >> 4;
    const int wm   = (wave & 1) * 64;
    const int wn   = (wave >> 1) * 32;
    const int n0   = blockIdx.x * 128;
    const int m0   = blockIdx.y * 128;

    v8f acc[4][2];
#pragma unroll
    for (int mi = 0; mi < 4; mi++)
#pragma unroll
        for (int ni = 0; ni < 2; ni++)
#pragma unroll
            for (int e = 0; e < 8; e++) acc[mi][ni][e] = 0.0f;

    for (int k0 = 0; k0 < DIM; k0 += 32) {
        if (k0 + 32 < DIM) {   // prefetch next K tile lines (global_prefetch_b8)
            __builtin_prefetch(inp + (size_t)(m0 + (t >> 1)) * DIM + (k0 + 32) + (t & 1) * 16, 0, 3);
            __builtin_prefetch(W + (size_t)(k0 + 32 + (t >> 3)) * NCOLS + n0 + (t & 7) * 16, 0, 3);
        }
        // ---- load A tile 128x32 f32 -> f16 LDS ----
#pragma unroll
        for (int i = 0; i < 4; i++) {
            const int idx = t + 256 * i;             // 0..1023
            const int row = idx >> 3;
            const int c4  = (idx & 7) * 4;
            const float4 v = *(const float4*)(inp + (size_t)(m0 + row) * DIM + k0 + c4);
            v4h hv;
            hv[0] = (_Float16)v.x; hv[1] = (_Float16)v.y;
            hv[2] = (_Float16)v.z; hv[3] = (_Float16)v.w;
            *(v4h*)&As[row * SA + c4] = hv;
        }
        // ---- load B tile 32x128, store transposed [n][k] ----
#pragma unroll
        for (int i = 0; i < 4; i++) {
            const int idx = t + 256 * i;             // 0..1023
            const int kr  = idx >> 5;                // 0..31
            const int c4  = (idx & 31) * 4;          // 0..124
            const float4 v = *(const float4*)(W + (size_t)(k0 + kr) * NCOLS + n0 + c4);
            Bs[(c4 + 0) * SA + kr] = (_Float16)v.x;
            Bs[(c4 + 1) * SA + kr] = (_Float16)v.y;
            Bs[(c4 + 2) * SA + kr] = (_Float16)v.z;
            Bs[(c4 + 3) * SA + kr] = (_Float16)v.w;
        }
        __syncthreads();

        FragU a[4], bf[2];
#pragma unroll
        for (int mi = 0; mi < 4; mi++) {
            const _Float16* p = &As[(wm + mi * 16 + r) * SA];
            a[mi].h[0] = *(const v8h*)(p + 8 * hi);        // K = 8*hi .. +7
            a[mi].h[1] = *(const v8h*)(p + 16 + 8 * hi);   // K = 16+8*hi .. +7
        }
#pragma unroll
        for (int ni = 0; ni < 2; ni++) {
            const _Float16* p = &Bs[(wn + ni * 16 + r) * SA];
            bf[ni].h[0] = *(const v8h*)(p + 8 * hi);       // K = 8*hi .. +7
            bf[ni].h[1] = *(const v8h*)(p + 16 + 8 * hi);  // K = 16+8*hi .. +7
        }
#pragma unroll
        for (int mi = 0; mi < 4; mi++)
#pragma unroll
            for (int ni = 0; ni < 2; ni++)
                acc[mi][ni] = __builtin_amdgcn_wmma_f32_16x16x32_f16(
                    false, a[mi].v, false, bf[ni].v, (short)0, acc[mi][ni], false, false);
        __syncthreads();
    }

    // ---- epilogue: +bias, RoPE, transpose 16x16 per wave, 16 B stores ----
    _Float16* tr = &Tr[wave * 256];
#pragma unroll
    for (int mi = 0; mi < 4; mi++) {
#pragma unroll
        for (int ni = 0; ni < 2; ni++) {
            const int ngBase = n0 + wn + ni * 16;        // subtile col base (mult of 16)
            const int ng     = ngBase + r;               // this lane's column
            const int d      = ng & 63;                  // head-dim index
            const float bv   = bias[ng];
            const float freq = __expf(-(float)(d & ~1) * LN10000_OVER_HS);
            const float sgn  = (d & 1) ? 1.0f : -1.0f;
#pragma unroll
            for (int rr = 0; rr < 8; rr++) {
                const int mg = m0 + wm + mi * 16 + rr + 8 * hi;
                const int s  = mg & (SEQ - 1);
                float x = acc[mi][ni][rr] + bv;
                float px = __shfl_xor(x, 1, 32);         // interleaved partner column
                const float ang = (float)s * freq;
                const float o = x * __cosf(ang) + sgn * px * __sinf(ang);
                tr[(rr + 8 * hi) * 16 + r] = (_Float16)o;
            }
            // subtile is entirely q or entirely k, one head, 16 consecutive d
            const int h   = ngBase >> 7;
            const int isK = (ngBase & 127) >> 6;
            const int d0  = ngBase & 63;
            const int mg2 = m0 + wm + mi * 16 + r;       // this lane writes row mg2
            const int s2  = mg2 & (SEQ - 1);
            const int b2  = mg2 >> 10;
            _Float16* base = isK ? kws : qws;
            _Float16* dst  = base + (((size_t)(b2 * N_HEADS + h) * SEQ + s2) * HEAD_SZ + d0 + hi * 8);
            *(v8h*)dst = *(const v8h*)&tr[r * 16 + hi * 8];
        }
    }
}

// ---------------------------------------------------------------------------
// Kernel 2: logits[bh][m][n] = (q[m]·k[n] - masks*NEG) / 8
// q-tile (128x64 f16) and k-tile staged into LDS by the Tensor Data Mover
// (tensor_load_to_lds, TENSORcnt), with D# row padding -> 72-half LDS stride.
// 8 waves x (4x2) WMMA subtiles; output via per-wave LDS transpose, b128 stores.
// ---------------------------------------------------------------------------
__global__ __launch_bounds__(256)
void logits_kernel(const _Float16* __restrict__ qws,
                   const _Float16* __restrict__ kws,
                   const int* __restrict__ mask,     // [8][1024]
                   float* __restrict__ out)          // [96][1024][1024]
{
    constexpr int SQ = 72;                           // padded LDS row stride (halves)
    __shared__ _Float16 qs[128 * SQ];                // 18432 B
    __shared__ _Float16 ks[128 * SQ];                // 18432 B
    __shared__ float    Tr[8 * 256];                 // 8192 B

    const int t    = threadIdx.x;
    const int lane = t & 31;
    const int wave = t >> 5;
    const int r    = lane & 15;
    const int hi   = lane >> 4;
    const int wm   = (wave & 1) * 64;
    const int wn   = (wave >> 1) * 32;
    const int n0   = blockIdx.x * 128;
    const int m0   = blockIdx.y * 128;
    const int bh   = blockIdx.z;
    const int b    = bh / N_HEADS;

    const _Float16* q = qws + (size_t)bh * SEQ * HEAD_SZ;
    const _Float16* k = kws + (size_t)bh * SEQ * HEAD_SZ;

    // ---- TDM: DMA both operand tiles into LDS ----
    if (wave == 0)
        tdm_load_2d_rows64(q + (size_t)m0 * HEAD_SZ, LDS_OFFSET(qs), 128);
    else if (wave == 1)
        tdm_load_2d_rows64(k + (size_t)n0 * HEAD_SZ, LDS_OFFSET(ks), 128);

    v8f acc[4][2];
#pragma unroll
    for (int mi = 0; mi < 4; mi++)
#pragma unroll
        for (int ni = 0; ni < 2; ni++)
#pragma unroll
            for (int e = 0; e < 8; e++) acc[mi][ni][e] = 0.0f;

    __builtin_amdgcn_s_wait_tensorcnt(0);
    __syncthreads();

#pragma unroll
    for (int kc = 0; kc < HEAD_SZ; kc += 32) {
        FragU a[4], bf[2];
#pragma unroll
        for (int mi = 0; mi < 4; mi++) {
            const _Float16* p = &qs[(wm + mi * 16 + r) * SQ + kc];
            a[mi].h[0] = *(const v8h*)(p + 8 * hi);
            a[mi].h[1] = *(const v8h*)(p + 16 + 8 * hi);
        }
#pragma unroll
        for (int ni = 0; ni < 2; ni++) {
            const _Float16* p = &ks[(wn + ni * 16 + r) * SQ + kc];
            bf[ni].h[0] = *(const v8h*)(p + 8 * hi);
            bf[ni].h[1] = *(const v8h*)(p + 16 + 8 * hi);
        }
#pragma unroll
        for (int mi = 0; mi < 4; mi++)
#pragma unroll
            for (int ni = 0; ni < 2; ni++)
                acc[mi][ni] = __builtin_amdgcn_wmma_f32_16x16x32_f16(
                    false, a[mi].v, false, bf[ni].v, (short)0, acc[mi][ni], false, false);
    }

    float* tr = &Tr[wave * 256];
#pragma unroll
    for (int mi = 0; mi < 4; mi++) {
#pragma unroll
        for (int ni = 0; ni < 2; ni++) {
            const int ng = n0 + wn + ni * 16 + r;            // key position (col)
            const float mn = 1.0f - (float)mask[b * SEQ + ng];
#pragma unroll
            for (int rr = 0; rr < 8; rr++) {
                const int mg = m0 + wm + mi * 16 + rr + 8 * hi;  // query position (row)
                const float mm = 1.0f - (float)mask[b * SEQ + mg];
                const float pen = NEG_BIG * (mn + mm + ((mg > ng) ? 1.0f : 0.0f));
                tr[(rr + 8 * hi) * 16 + r] = (acc[mi][ni][rr] - pen) * 0.125f;
            }
            const int mg2 = m0 + wm + mi * 16 + r;
            float* dst = out + ((size_t)bh * SEQ + mg2) * SEQ + (n0 + wn + ni * 16 + hi * 8);
            const float4 v0 = *(const float4*)&tr[r * 16 + hi * 8];
            const float4 v1 = *(const float4*)&tr[r * 16 + hi * 8 + 4];
            *(float4*)dst       = v0;
            *(float4*)(dst + 4) = v1;
        }
    }
}

// ---------------------------------------------------------------------------
extern "C" void kernel_launch(void* const* d_in, const int* in_sizes, int n_in,
                              void* d_out, int out_size, void* d_ws, size_t ws_size,
                              hipStream_t stream) {
    const float* inputs = (const float*)d_in[0];   // [8,1024,1024] f32
    const float* W      = (const float*)d_in[1];   // [1024,1536]   f32
    const float* bias   = (const float*)d_in[2];   // [1536]        f32
    const int*   mask   = (const int*)  d_in[3];   // [8,1024]      i32
    float*       out    = (float*)d_out;           // [8,12,1024,1024] f32

    _Float16* qws = (_Float16*)d_ws;               // 96*1024*64 halves
    _Float16* kws = qws + (size_t)BATCH * N_HEADS * SEQ * HEAD_SZ;

    // GEMM1 + RoPE: grid = (N tiles 1536/128, M tiles 8192/128)
    proj_rope_kernel<<<dim3(NCOLS / 128, (BATCH * SEQ) / 128, 1), 256, 0, stream>>>(
        inputs, W, bias, qws, kws);

    // GEMM2 + mask: grid = (n tiles, m tiles, b*h)
    logits_kernel<<<dim3(SEQ / 128, SEQ / 128, BATCH * N_HEADS), 256, 0, stream>>>(
        qws, kws, mask, out);
}
